// LSTMLayer_16381005267364
// MI455X (gfx1250) — compile-verified, matrix operand order checked by reading, not run on hardware
//
#include <hip/hip_runtime.h>

// ---------------------------------------------------------------------------
// LSTM layer for MI455X (gfx1250, wave32, WMMA, TDM).
// Phase 1: Z = [x @ Wx_i^T | x @ Wx_f^T | x @ Wx_o^T | x @ Wx_c^T] + b
//          bf16 WMMA GEMM (M=65536, K=512, N=2048), f32 accumulate,
//          double-buffered LDS tiles (1 barrier per K chunk).
// Phase 2: persistent-kernel recurrence, 32 WGs, Wh slices DMA'd into LDS by
//          the Tensor Data Mover, device-scope atomic grid barrier per step.
// ---------------------------------------------------------------------------

#define T_STEPS 2048
#define BATCH   32
#define I_DIM   512
#define H_DIM   512
#define NGATE   4
#define NCOL    (NGATE * H_DIM)     // 2048
#define MROWS   (T_STEPS * BATCH)   // 65536

typedef __attribute__((ext_vector_type(16))) __bf16         v16bf;
typedef __attribute__((ext_vector_type(8)))  __bf16         v8bf;
typedef __attribute__((ext_vector_type(8)))  float          v8f;
typedef __attribute__((ext_vector_type(8)))  unsigned short us8;
typedef __attribute__((ext_vector_type(4)))  unsigned int   u32x4;
typedef __attribute__((ext_vector_type(8)))  unsigned int   u32x8;
typedef __attribute__((ext_vector_type(4)))  int            i32x4;
typedef __attribute__((ext_vector_type(8)))  int            i32x8;

static __device__ __forceinline__ unsigned short f32_to_bf16(float f) {
  unsigned u = __builtin_bit_cast(unsigned, f);
  u += 0x7FFFu + ((u >> 16) & 1u);        // round to nearest even
  return (unsigned short)(u >> 16);
}

static __device__ __forceinline__ float sigmoidf_fast(float x) {
  return 1.0f / (1.0f + __expf(-x));
}

// Load one 16x32 bf16 fragment (A or B operand of v_wmma_f32_16x16x32_bf16)
// from a row-major tile (16 rows of `stride` bf16 elements) at `base`.
// Layout per CDNA5 ISA 7.12.2:
//   lanes 0-15  : row = lane,     K 0-7 (VGPR0-3), K 16-23 (VGPR4-7)
//   lanes 16-31 : row = lane-16,  K 8-15,          K 24-31
static __device__ __forceinline__ v16bf load_frag_16x32(
    const unsigned short* base, int stride) {
  const int lane = threadIdx.x & 31;
  const int r    = lane & 15;
  const int hf   = lane >> 4;
  const unsigned short* p = base + r * stride + hf * 8;
  v8bf lo = *(const v8bf*)(p);
  v8bf hi = *(const v8bf*)(p + 16);
  v16bf out;
#pragma unroll
  for (int i = 0; i < 8; ++i) { out[i] = lo[i]; out[8 + i] = hi[i]; }
  return out;
}

// TDM: DMA a contiguous 64KB region (64 rows x 512 bf16) from global memory
// into LDS offset 0 (this kernel has no static LDS, so dynamic LDS base = 0).
// D# per CDNA5 ISA: 2D tensor, data_size=8B units,
// tensor_dim0 = tile_dim0 = 8192 units, tensor_dim0_stride = 8192,
// tensor_dim1 = tile_dim1 = 1, group0.type=2, count=1, no multicast.
static __device__ __forceinline__ void tdm_load_lds0_64k(const void* gsrc) {
  unsigned long long ga = (unsigned long long)gsrc;
  u32x4 g0 = { 1u,                                   // count=1, user mode
               0u,                                   // lds_addr = 0
               (unsigned)(ga & 0xFFFFFFFFu),         // global_addr[31:0]
               (unsigned)((ga >> 32) & 0x01FFFFFFu)  // global_addr[56:32]
                   | 0x80000000u };                  // type = 2 ("image")
  u32x8 g1u = { 0x00030000u,   // wg_mask=0, data_size=3 (8B), no flags
                0x20000000u,   // atomic_bar_addr=0, tensor_dim0.lo16=8192
                0x00010000u,   // tensor_dim0.hi16=0, tensor_dim1.lo16=1
                0x20000000u,   // tensor_dim1.hi16=0, tile_dim0=8192
                0x00000001u,   // tile_dim1=1, tile_dim2=0
                8192u,         // tensor_dim0_stride[31:0]
                0u,            // stride0.hi16=0, tensor_dim1_stride.lo16=0
                0u };          // tensor_dim1_stride[47:16]=0
  i32x8 g1 = __builtin_bit_cast(i32x8, g1u);
  i32x4 z4 = {0, 0, 0, 0};
#if __clang_major__ >= 23
  i32x8 z8 = {0, 0, 0, 0, 0, 0, 0, 0};
  __builtin_amdgcn_tensor_load_to_lds(g0, g1, z4, z4, z8, 0);
#else
  __builtin_amdgcn_tensor_load_to_lds(g0, g1, z4, z4, 0);
#endif
}

// --------------------------- prep kernels ----------------------------------

__global__ void k_f32_to_bf16(const float* __restrict__ src,
                              unsigned short* __restrict__ dst,
                              unsigned int n) {
  unsigned int idx = blockIdx.x * 256u + threadIdx.x;
  if (idx < n) dst[idx] = f32_to_bf16(src[idx]);
}

// Wh packing for phase 2: WG j owns hidden units n in [16j, 16j+16) for all
// 4 gates.  Row ((n>>4)*64 + g*16 + (n&15)) of the packed matrix = row n of
// W_h{g}.  K (512) stays contiguous.
__global__ void k_pack_wh(const float* __restrict__ src,
                          unsigned short* __restrict__ dst, int g) {
  unsigned int idx = blockIdx.x * 256u + threadIdx.x;
  if (idx < (unsigned)H_DIM * H_DIM) {
    int n = idx >> 9;
    int k = idx & 511;
    dst[((size_t)((n >> 4) * 64 + g * 16 + (n & 15))) * H_DIM + k] =
        f32_to_bf16(src[idx]);
  }
}

__global__ void k_pack_bias(const float* __restrict__ bi,
                            const float* __restrict__ bf,
                            const float* __restrict__ bo,
                            const float* __restrict__ bc,
                            float* __restrict__ dst) {
  int idx = blockIdx.x * 256 + threadIdx.x;   // 0..2047, col = g*512 + n
  if (idx < NCOL) {
    int g = idx >> 9, n = idx & 511;
    const float* s = (g == 0) ? bi : (g == 1) ? bf : (g == 2) ? bo : bc;
    dst[idx] = s[n];
  }
}

// --------------------------- phase 1: input GEMM ---------------------------
// C[M=65536, N=2048] = Xb[M,512] * Wxc[N,512]^T + bias.  64x64 tile per WG,
// 8 waves: wave w -> (mp = w>>2) covers rows 32*mp..+32 (two 16-row tiles),
// (nt = w&3) covers cols 16*nt..+16.  LDS tiles are double-buffered so each
// 32-wide K chunk costs one workgroup barrier and the next chunk's
// global_load_b128s overlap the WMMAs.
__global__ void __launch_bounds__(256) k_gemm_in(
    const unsigned short* __restrict__ xb,
    const unsigned short* __restrict__ wxc,
    const float* __restrict__ bcat,
    float* __restrict__ Z) {
  __shared__ __align__(16) unsigned short At[2][64 * 32];
  __shared__ __align__(16) unsigned short Bt[2][64 * 32];

  const int tid = threadIdx.x;
  const int Mb  = blockIdx.y * 64;
  const int Nb  = blockIdx.x * 64;
  const int wave = tid >> 5;
  const int mp   = wave >> 2;       // 0..1
  const int nt   = wave & 3;        // 0..3
  const int ldrow = tid >> 2;       // 0..63
  const int ldseg = (tid & 3) * 8;  // 0,8,16,24 (bf16 elems)

  const unsigned short* xrow = &xb[(size_t)(Mb + ldrow) * I_DIM + ldseg];
  const unsigned short* wrow = &wxc[(size_t)(Nb + ldrow) * I_DIM + ldseg];

  auto stage = [&](int kc, int buf) {
    us8 xa = *(const us8*)(xrow + kc * 32);
    us8 wa = *(const us8*)(wrow + kc * 32);
    *(us8*)&At[buf][ldrow * 32 + ldseg] = xa;
    *(us8*)&Bt[buf][ldrow * 32 + ldseg] = wa;
  };

  v8f acc0 = {};
  v8f acc1 = {};
  const int NK = I_DIM / 32;

  stage(0, 0);
  __syncthreads();

  for (int kc = 0; kc < NK; ++kc) {
    const int cur = kc & 1;
    if (kc + 1 < NK) {
      stage(kc + 1, cur ^ 1);                          // overlap with WMMA
      __builtin_prefetch(xrow + (kc + 2) * 32, 0, 3);  // global_prefetch_b8
      __builtin_prefetch(wrow + (kc + 2) * 32, 0, 3);
    }
    v16bf a0 = load_frag_16x32(&At[cur][(mp * 32 + 0) * 32], 32);
    v16bf a1 = load_frag_16x32(&At[cur][(mp * 32 + 16) * 32], 32);
    v16bf b  = load_frag_16x32(&Bt[cur][(nt * 16) * 32], 32);
    acc0 = __builtin_amdgcn_wmma_f32_16x16x32_bf16(false, a0, false, b,
                                                   (short)0, acc0, false, false);
    acc1 = __builtin_amdgcn_wmma_f32_16x16x32_bf16(false, a1, false, b,
                                                   (short)0, acc1, false, false);
    __syncthreads();
  }

  // C/D layout: VGPR v -> row v + 8*(lane>=16); col = lane%16
  const int lane = tid & 31, r = lane & 15, hf = lane >> 4;
  const int col = Nb + nt * 16 + r;
  const float bias = bcat[col];
#pragma unroll
  for (int v = 0; v < 8; ++v) {
    int row = Mb + mp * 32 + v + 8 * hf;
    Z[(size_t)row * NCOL + col]        = acc0[v] + bias;
    Z[(size_t)(row + 16) * NCOL + col] = acc1[v] + bias;
  }
}

// --------------------------- phase 2: recurrence ---------------------------
// 32 persistent WGs x 256 threads.  WG j owns hidden cols [16j,16j+16) for
// all 4 gates.  Its Wh slice (64x512 bf16 = 64KB) is DMA'd into LDS once by
// the TDM and reused for all 2048 steps.  Per step: stage h_prev (bf16) to
// LDS, 8 waves each do one 16x16 WMMA output tile (wave = batch-half x gate),
// add Z[t], f32 gate math, write h slice, device-scope atomic grid barrier.
__global__ void __launch_bounds__(256) k_lstm_rec(
    const unsigned short* __restrict__ whc,
    const float* __restrict__ Z,
    float* __restrict__ out,            // y[T,B,H] ++ hT[B,H] ++ cT[B,H]
    unsigned short* __restrict__ hbuf,  // 2 ping-pong B*H bf16 buffers
    unsigned int* __restrict__ bar) {
  extern __shared__ char smem[];
  // WhS MUST stay at dynamic-LDS offset 0 (TDM descriptor hardcodes it).
  unsigned short* WhS = (unsigned short*)smem;                       // 64x512 bf16
  unsigned short* hS  = (unsigned short*)(smem + 64 * 512 * 2);      // 32x512 bf16
  float* zS = (float*)(smem + 64 * 512 * 2 + 32 * 512 * 2);          // 32x64 f32
  float* cS = (float*)(smem + 64 * 512 * 2 + 32 * 512 * 2 + 32 * 64 * 4); // 32x16

  const int tid  = threadIdx.x;
  const int j    = blockIdx.x;     // hidden slice 0..31
  const int wave = tid >> 5;
  const int mh   = wave >> 2;      // batch half 0..1
  const int gate = wave & 3;       // i,f,o,c

  // One TDM DMA pulls this WG's 64KB Wh slice into LDS (wave 0 issues it and
  // waits on TENSORcnt; the barrier publishes it to the other waves).
  if (wave == 0) {
    tdm_load_lds0_64k(&whc[(size_t)j * 64 * 512]);
    __builtin_amdgcn_s_wait_tensorcnt(0);
  }
  __syncthreads();

  for (int t = 0; t < T_STEPS; ++t) {
    if (t > 0) {
      // Stage full h_{t-1} (bf16) into LDS, then GEMM.
      const unsigned short* hprev = hbuf + ((t - 1) & 1) * (BATCH * H_DIM);
      for (int i = tid; i < BATCH * H_DIM / 8; i += 256)
        *(us8*)&hS[i * 8] = *(const us8*)&hprev[i * 8];
      __syncthreads();

      v8f acc = {};
#pragma unroll
      for (int kc = 0; kc < H_DIM / 32; ++kc) {
        v16bf a = load_frag_16x32(&hS[(mh * 16) * H_DIM + kc * 32], H_DIM);
        v16bf b = load_frag_16x32(&WhS[(gate * 16) * H_DIM + kc * 32], H_DIM);
        acc = __builtin_amdgcn_wmma_f32_16x16x32_bf16(false, a, false, b,
                                                      (short)0, acc, false, false);
      }
      const int lane = tid & 31, r = lane & 15, hf = lane >> 4;
      const size_t zrow0 = (size_t)t * BATCH;
#pragma unroll
      for (int v = 0; v < 8; ++v) {
        int m = mh * 16 + v + 8 * hf;  // batch index
        float z = acc[v] +
                  Z[(zrow0 + m) * NCOL + (size_t)gate * H_DIM + j * 16 + r];
        zS[m * 64 + gate * 16 + r] = z;
      }
      __syncthreads();
    }

    // Elementwise gates + state update: 512 (b,s) slots, 2 per thread.
#pragma unroll
    for (int s2 = 0; s2 < 2; ++s2) {
      int slot = tid + s2 * 256;
      int b = slot >> 4;
      int s = slot & 15;
      float zi, zf, zo, zc;
      if (t == 0) {
        size_t zr = (size_t)b * NCOL + j * 16 + s;   // row t=0
        zi = Z[zr + 0 * H_DIM];
        zf = 0.0f;
        zo = Z[zr + 2 * H_DIM];
        zc = Z[zr + 3 * H_DIM];
      } else {
        zi = zS[b * 64 + 0 * 16 + s];
        zf = zS[b * 64 + 1 * 16 + s];
        zo = zS[b * 64 + 2 * 16 + s];
        zc = zS[b * 64 + 3 * 16 + s];
      }
      float ig = sigmoidf_fast(zi);
      float og = sigmoidf_fast(zo);
      float gg = tanhf(zc);
      float c;
      if (t == 0) {
        c = ig * gg;
      } else {
        float fg = sigmoidf_fast(zf);
        c = fg * cS[b * 16 + s] + ig * gg;
      }
      cS[b * 16 + s] = c;
      float h = og * tanhf(c);

      int col = j * 16 + s;
      out[((size_t)t * BATCH + b) * H_DIM + col] = h;                 // y
      hbuf[(t & 1) * (BATCH * H_DIM) + b * H_DIM + col] = f32_to_bf16(h);
      if (t == T_STEPS - 1) {
        size_t ylen = (size_t)T_STEPS * BATCH * H_DIM;
        out[ylen + (size_t)b * H_DIM + col] = h;                      // hT
        out[ylen + BATCH * H_DIM + (size_t)b * H_DIM + col] = c;      // cT
      }
    }

    __threadfence();      // make this WG's h-slice stores device-visible
    __syncthreads();

    if (t + 1 < T_STEPS) {
      if (tid == 0)
        __hip_atomic_fetch_add(bar, 1u, __ATOMIC_RELEASE,
                               __HIP_MEMORY_SCOPE_AGENT);
      const unsigned int target = 32u * (unsigned)(t + 1);
      while (__hip_atomic_load(bar, __ATOMIC_ACQUIRE,
                               __HIP_MEMORY_SCOPE_AGENT) < target)
        __builtin_amdgcn_s_sleep(2);
    }
  }
}

// --------------------------- host launch -----------------------------------

extern "C" void kernel_launch(void* const* d_in, const int* in_sizes, int n_in,
                              void* d_out, int out_size, void* d_ws,
                              size_t ws_size, hipStream_t stream) {
  const float* x    = (const float*)d_in[0];
  const float* W_xi = (const float*)d_in[1];
  const float* b_xi = (const float*)d_in[2];
  const float* W_hi = (const float*)d_in[3];
  const float* W_xf = (const float*)d_in[4];
  const float* b_xf = (const float*)d_in[5];
  const float* W_hf = (const float*)d_in[6];
  const float* W_xo = (const float*)d_in[7];
  const float* b_xo = (const float*)d_in[8];
  const float* W_ho = (const float*)d_in[9];
  const float* W_xc = (const float*)d_in[10];
  const float* b_xc = (const float*)d_in[11];
  const float* W_hc = (const float*)d_in[12];

  // Workspace carve-up (~580MB; Z dominates at 512MB f32).
  char* ws = (char*)d_ws;
  size_t off = 0;
  auto carve = [&](size_t bytes) {
    char* p = ws + off;
    off += (bytes + 255) & ~(size_t)255;
    return p;
  };
  unsigned short* xb   = (unsigned short*)carve((size_t)MROWS * I_DIM * 2);
  unsigned short* wxc  = (unsigned short*)carve((size_t)NCOL * I_DIM * 2);
  unsigned short* whc  = (unsigned short*)carve((size_t)NCOL * H_DIM * 2);
  float*          bcat = (float*)carve(NCOL * sizeof(float));
  float*          Z    = (float*)carve((size_t)MROWS * NCOL * sizeof(float));
  unsigned short* hbuf = (unsigned short*)carve(2 * BATCH * H_DIM * 2);
  unsigned int*   bar  = (unsigned int*)carve(256);

  hipMemsetAsync(bar, 0, sizeof(unsigned int), stream);

  // fp32 -> bf16 conversions / packing.
  {
    unsigned int n = (unsigned)MROWS * I_DIM;
    k_f32_to_bf16<<<dim3((n + 255) / 256), dim3(256), 0, stream>>>(x, xb, n);
  }
  const float* wx[4] = {W_xi, W_xf, W_xo, W_xc};
  const float* wh[4] = {W_hi, W_hf, W_ho, W_hc};
  for (int g = 0; g < 4; ++g) {
    unsigned int n = (unsigned)H_DIM * I_DIM;
    k_f32_to_bf16<<<dim3((n + 255) / 256), dim3(256), 0, stream>>>(
        wx[g], wxc + (size_t)g * H_DIM * I_DIM, n);
    k_pack_wh<<<dim3((n + 255) / 256), dim3(256), 0, stream>>>(wh[g], whc, g);
  }
  k_pack_bias<<<dim3((NCOL + 255) / 256), dim3(256), 0, stream>>>(
      b_xi, b_xf, b_xo, b_xc, bcat);

  // Phase 1: input-projection GEMM (32768 WGs).
  k_gemm_in<<<dim3(NCOL / 64, MROWS / 64), dim3(256), 0, stream>>>(xb, wxc,
                                                                   bcat, Z);

  // Phase 2: persistent recurrent kernel, 32 resident WGs, ~106KB LDS each.
  size_t lds = (size_t)64 * 512 * 2 + 32 * 512 * 2 + 32 * 64 * 4 + 32 * 16 * 4;
  k_lstm_rec<<<dim3(32), dim3(256), lds, stream>>>(whc, Z, (float*)d_out,
                                                   hbuf, bar);
}
